// GNN_GIN_30915174597111
// MI455X (gfx1250) — compile-verified, matrix operand order checked by reading, not run on hardware
//
#include <hip/hip_runtime.h>

// ---------------------------------------------------------------------------
// GIN GNN forward for MI455X (gfx1250).  Activations are kept in bf16
// end-to-end between GEMMs; all dense GEMMs go through
// v_wmma_f32_16x16x32_bf16 (bf16 in, f32 accumulate), 4 WMMAs per wave per
// k-step (32x32 register tile).  The GEMM inner loop is pure data movement:
// global b128 -> LDS b128 -> ds_load_b128 fragments -> WMMA (no conversion
// VALU between barriers).  Sparse aggregation + pooling use
// global_atomic_add_f32.  wave32 throughout.
// ---------------------------------------------------------------------------

#define N_NODES_C  50000
#define N_EDGES_C  800000
#define N_GRAPHS_C 64

typedef __bf16 bf16x16 __attribute__((ext_vector_type(16)));
typedef float  floatx8 __attribute__((ext_vector_type(8)));

enum { ACT_NONE = 0, ACT_SIGMOID = 1, ACT_RELU = 2, ACT_ELU = 3, ACT_SELU = 4, ACT_GELU = 5 };

__device__ __forceinline__ float apply_act(float v, int act) {
    switch (act) {
    case ACT_SIGMOID: return 1.0f / (1.0f + expf(-v));
    case ACT_RELU:    return v > 0.0f ? v : 0.0f;
    case ACT_ELU:     return v > 0.0f ? v : expm1f(v);
    case ACT_SELU: {
        const float a = 1.6732632423543772f, s = 1.0507009873554805f;
        return v > 0.0f ? s * v : s * a * expm1f(v);
    }
    case ACT_GELU:    return 0.5f * v * (1.0f + erff(v * 0.7071067811865476f));
    default:          return v;
    }
}

// round-to-nearest-even f32 -> bf16 (stored as ushort)
__device__ __forceinline__ unsigned short f2bf(float f) {
    unsigned int u = __float_as_uint(f);
    u += 0x7FFFu + ((u >> 16) & 1u);
    return (unsigned short)(u >> 16);
}

// --------------------------- small utility kernels -------------------------

__global__ void k_cvt_bf16(const float* __restrict__ src, unsigned short* __restrict__ dst, int n) {
    int i = blockIdx.x * blockDim.x + threadIdx.x;
    if (i < n) dst[i] = f2bf(src[i]);
}

__global__ void k_zero_f32(float* __restrict__ p, size_t n) {
    size_t i = (size_t)blockIdx.x * blockDim.x + threadIdx.x;
    if (i < n) p[i] = 0.0f;
}

// agg[dst] += x[src], vectorized float4 over feature dim (D multiple of 4)
__global__ void k_edge_scatter(const float* __restrict__ X, const int* __restrict__ ei,
                               float* __restrict__ Agg, int E, int D) {
    int vecs = D >> 2;
    long long t = (long long)blockIdx.x * blockDim.x + threadIdx.x;
    if (t >= (long long)E * vecs) return;
    int e = (int)(t / vecs);
    int c = (int)(t % vecs);
    int src = ei[e];
    int dst = ei[E + e];
    float4 v = ((const float4*)(X + (size_t)src * D))[c];
    float* base = Agg + (size_t)dst * D + (size_t)c * 4;
    atomicAdd(base + 0, v.x);
    atomicAdd(base + 1, v.y);
    atomicAdd(base + 2, v.z);
    atomicAdd(base + 3, v.w);
}

// Yb = bf16((1+eps)*X + Agg)   (feeds the next GEMM directly as bf16 A)
__global__ void k_gin_combine_bf(const float* __restrict__ X, const float* __restrict__ Agg,
                                 const float* __restrict__ epsp,
                                 unsigned short* __restrict__ Yb, size_t n) {
    size_t i = (size_t)blockIdx.x * blockDim.x + threadIdx.x;
    if (i < n) Yb[i] = f2bf((1.0f + epsp[0]) * X[i] + Agg[i]);
}

// one block per feature: biased mean/var over N rows
__global__ void k_bn_stats(const float* __restrict__ X, float* __restrict__ meanvar, int N, int D) {
    int f = blockIdx.x;
    float s = 0.0f, s2 = 0.0f;
    for (int i = threadIdx.x; i < N; i += blockDim.x) {
        float v = X[(size_t)i * D + f];
        s += v; s2 += v * v;
    }
    __shared__ float sh[256], sh2[256];
    sh[threadIdx.x] = s; sh2[threadIdx.x] = s2;
    __syncthreads();
    for (int off = 128; off > 0; off >>= 1) {
        if ((int)threadIdx.x < off) {
            sh[threadIdx.x]  += sh[threadIdx.x + off];
            sh2[threadIdx.x] += sh2[threadIdx.x + off];
        }
        __syncthreads();
    }
    if (threadIdx.x == 0) {
        float mean = sh[0] / (float)N;
        float var  = sh2[0] / (float)N - mean * mean;
        meanvar[f]     = mean;
        meanvar[D + f] = var;
    }
}

__global__ void k_bn_apply_act(const float* __restrict__ X, const float* __restrict__ meanvar,
                               const float* __restrict__ g, const float* __restrict__ b,
                               float* __restrict__ Y, size_t n, int D, int act) {
    size_t i = (size_t)blockIdx.x * blockDim.x + threadIdx.x;
    if (i >= n) return;
    int f = (int)(i % D);
    float mean = meanvar[f], var = meanvar[D + f];
    float v = (X[i] - mean) * rsqrtf(var + 1e-5f) * g[f] + b[f];
    Y[i] = apply_act(v, act);
}

// G[batch[n], colOff + f] += X[n, f]
__global__ void k_pool(const float* __restrict__ X, const int* __restrict__ batch,
                       float* __restrict__ G, int N, int D, int colOff, int GD) {
    long long i = (long long)blockIdx.x * blockDim.x + threadIdx.x;
    if (i >= (long long)N * D) return;
    int n = (int)(i / D), f = (int)(i % D);
    atomicAdd(&G[(size_t)batch[n] * GD + colOff + f], X[i]);
}

// tiny GEMM for the 64-row head: one thread per output element
__global__ void k_small_gemm(const float* __restrict__ A, const float* __restrict__ W,
                             const float* __restrict__ bias, float* __restrict__ C,
                             int N, int K, int M, int act) {
    int idx = blockIdx.x * blockDim.x + threadIdx.x;
    if (idx >= N * M) return;
    int row = idx / M, col = idx % M;
    float s = bias[col];
    for (int k = 0; k < K; ++k) s += A[(size_t)row * K + k] * W[(size_t)k * M + col];
    C[idx] = apply_act(s, act);
}

// ----------------------------- WMMA GEMM kernel ----------------------------
// out = act(Abf[N,K] (bf16) @ Wbf[K,M] (bf16) + bias), K % 32 == 0, M % 4 == 0.
// Output either f32 (Cf) or bf16 (Cb) per outBf flag.
// 256 threads = 8 waves in a 4x2 grid.  Workgroup tile: 128 rows x 64 cols.
// Each wave computes a 32x32 register tile (4 accumulators, fragment reuse).
#define GT_R 128
#define GT_C 64
#define GK   32
#define STR  40   // ushort stride: 80 B rows -> b128-aligned, staggered banks

__device__ __forceinline__ void store_tile(const floatx8& acc,
                                           float* __restrict__ Cf, unsigned short* __restrict__ Cb,
                                           const float* __restrict__ bias,
                                           int N, int M, int row0, int col, int act, int outBf) {
    if (col >= M) return;
    float bv = bias[col];
    #pragma unroll
    for (int r = 0; r < 8; ++r) {
        int row = row0 + r;
        if (row < N) {
            float o = apply_act(acc[r] + bv, act);
            if (outBf) Cb[(size_t)row * M + col] = f2bf(o);
            else       Cf[(size_t)row * M + col] = o;
        }
    }
}

__global__ __launch_bounds__(256)
void k_gemm_bf16_wmma(const unsigned short* __restrict__ Abf,
                      const unsigned short* __restrict__ Wbf,
                      const float* __restrict__ bias,
                      float* __restrict__ Cf, unsigned short* __restrict__ Cb,
                      int N, int K, int M, int act, int outBf) {
    __shared__ __align__(16) unsigned short As[GT_R * STR]; // 10240 B
    __shared__ __align__(16) unsigned short Bs[GT_C * STR]; //  5120 B

    const int tid  = threadIdx.x;
    const int lane = tid & 31;
    const int w    = tid >> 5;
    const int rowBase = blockIdx.x * GT_R;
    const int colBase = blockIdx.y * GT_C;
    const int wr = (w & 3) * 32;   // wave row offset inside tile
    const int wc = (w >> 2) * 32;  // wave col offset inside tile

    const int ml   = lane & 15;    // M (or N) index held by this lane
    const int half = lane >> 4;    // K-group select per ISA 16-bit layout

    // per-thread staging coordinates (reused every k-step)
    const int ar  = tid >> 2;              // A row (first of 2 strided by 64)
    const int ak8 = (tid & 3) << 3;        // A k-octet (8 bf16 = 16 B)
    const int bk  = tid >> 4;              // B k (first of 2 strided by 16)
    const int bc4 = (tid & 15) << 2;       // B col quad

    floatx8 acc00 = {}, acc01 = {}, acc10 = {}, acc11 = {};

    for (int k0 = 0; k0 < K; k0 += GK) {
        // ---- stage A tile (128 rows x 32 k): pure b128 copy, branch-free guard ----
        #pragma unroll
        for (int it = 0; it < 2; ++it) {
            int r  = ar + it * 64;
            int gr = rowBase + r;
            int rs = gr < N ? gr : N - 1;
            uint4 v = *(const uint4*)(Abf + (size_t)rs * K + (k0 + ak8));
            if (gr >= N) { v.x = 0u; v.y = 0u; v.z = 0u; v.w = 0u; }
            *(uint4*)&As[r * STR + ak8] = v;     // 16B-aligned: 80r + 16*(ak8/8)
        }
        // ---- stage W^T tile (64 cols x 32 k): b64 loads coalesced along cols ----
        #pragma unroll
        for (int it = 0; it < 2; ++it) {
            int k  = bk + it * 16;
            int gc = colBase + bc4;
            uint2 wv;
            if (gc < M) {                // M%4==0 -> all four lanes in range
                wv = *(const uint2*)(Wbf + (size_t)(k0 + k) * M + gc);
            } else { wv.x = 0u; wv.y = 0u; }
            Bs[(bc4 + 0) * STR + k] = (unsigned short)(wv.x & 0xFFFFu);
            Bs[(bc4 + 1) * STR + k] = (unsigned short)(wv.x >> 16);
            Bs[(bc4 + 2) * STR + k] = (unsigned short)(wv.y & 0xFFFFu);
            Bs[(bc4 + 3) * STR + k] = (unsigned short)(wv.y >> 16);
        }
        // ---- prefetch next k-step while this one computes ----
        if (k0 + GK < K) {
            int grn = rowBase + ar;
            int rsn = grn < N ? grn : N - 1;
            __builtin_prefetch(Abf + (size_t)rsn * K + (k0 + GK + ak8), 0, 1);
            int gcn = colBase + bc4;
            int csn = gcn < M ? gcn : 0;
            __builtin_prefetch(Wbf + (size_t)(k0 + GK + bk) * M + csn, 0, 1);
        }
        __syncthreads();

        // ---- build fragments per ISA §7.12.2 16-bit 16x32 layout ----
        //  lane<16: VGPR j -> K = 2j..2j+1 (j<4) / 16+2(j-4).. (j>=4); lane>=16: +8
        union { bf16x16 v; unsigned int u[8]; } a0, a1, b0, b1;
        #pragma unroll
        for (int j = 0; j < 8; ++j) {
            int kb = (j < 4) ? (half * 8 + 2 * j) : (16 + half * 8 + 2 * (j - 4));
            a0.u[j] = *(const unsigned int*)&As[(wr + ml) * STR + kb];
            a1.u[j] = *(const unsigned int*)&As[(wr + 16 + ml) * STR + kb];
            b0.u[j] = *(const unsigned int*)&Bs[(wc + ml) * STR + kb];
            b1.u[j] = *(const unsigned int*)&Bs[(wc + 16 + ml) * STR + kb];
        }
        acc00 = __builtin_amdgcn_wmma_f32_16x16x32_bf16(false, a0.v, false, b0.v,
                                                        (short)0, acc00, false, false);
        acc01 = __builtin_amdgcn_wmma_f32_16x16x32_bf16(false, a0.v, false, b1.v,
                                                        (short)0, acc01, false, false);
        acc10 = __builtin_amdgcn_wmma_f32_16x16x32_bf16(false, a1.v, false, b0.v,
                                                        (short)0, acc10, false, false);
        acc11 = __builtin_amdgcn_wmma_f32_16x16x32_bf16(false, a1.v, false, b1.v,
                                                        (short)0, acc11, false, false);
        __syncthreads();
    }

    // ---- epilogue: C/D layout — VGPR r: lanes0-15 M=r, lanes16-31 M=8+r ----
    const int r00 = rowBase + wr + 8 * half;
    const int r10 = r00 + 16;
    const int c0  = colBase + wc + ml;
    const int c1  = c0 + 16;
    store_tile(acc00, Cf, Cb, bias, N, M, r00, c0, act, outBf);
    store_tile(acc01, Cf, Cb, bias, N, M, r00, c1, act, outBf);
    store_tile(acc10, Cf, Cb, bias, N, M, r10, c0, act, outBf);
    store_tile(acc11, Cf, Cb, bias, N, M, r10, c1, act, outBf);
}

// ------------------------------ host driver --------------------------------

// d_in flattening assumption: setup_inputs() dict insertion order, recursively.
enum {
    IN_X = 0, IN_EI = 1, IN_BATCH = 2,
    IN_LIN1_W = 3, IN_LIN1_B = 4, IN_LIN2_W = 5, IN_LIN2_B = 6,
    IN_MLP1 = 7, IN_MLP2 = 17, IN_MLP3 = 27,   // each: il.w,il.b,hl1.w,hl1.b,hl2.w,hl2.b,hl3.w,hl3.b,ol.w,ol.b
    IN_EPS1 = 37, IN_EPS2 = 38, IN_EPS3 = 39,
    IN_BN1_G = 40, IN_BN1_B = 41, IN_BN2_G = 42, IN_BN2_B = 43, IN_BN3_G = 44, IN_BN3_B = 45,
    IN_HEAD = 46                               // il.w,il.b,hl1.w,hl1.b,hl2.w,hl2.b,hl3.w,hl3.b,hl4.w,hl4.b,ol.w,ol.b
};

extern "C" void kernel_launch(void* const* d_in, const int* in_sizes, int n_in,
                              void* d_out, int out_size, void* d_ws, size_t ws_size,
                              hipStream_t stream) {
    (void)in_sizes; (void)n_in; (void)out_size; (void)ws_size;
    const int NN = N_NODES_C, EE = N_EDGES_C, NG = N_GRAPHS_C;

    const float* x     = (const float*)d_in[IN_X];
    const int*   ei    = (const int*)d_in[IN_EI];
    const int*   batch = (const int*)d_in[IN_BATCH];
    auto F = [&](int i) { return (const float*)d_in[i]; };

    // ---- workspace layout (float units; bf16 regions counted as /2) ----
    float* ws = (float*)d_ws;
    size_t off = 0;
    float* Hf = ws + off; off += (size_t)NN * 128;   // f32 h (lin2 out) / agg input
    float* AG = ws + off; off += (size_t)NN * 128;   // f32 aggregation buffer
    float* Of = ws + off; off += (size_t)NN * 32;    // f32 MLP final output (BN input)
    float* X1 = ws + off; off += (size_t)NN * 32;
    float* X2 = ws + off; off += (size_t)NN * 32;
    float* X3 = ws + off; off += (size_t)NN * 32;
    float* G  = ws + off; off += (size_t)NG * 96;
    float* MV = ws + off; off += 64;                 // mean[32] + var[32]
    float* H0 = ws + off; off += (size_t)NG * 512;
    float* H1 = ws + off; off += (size_t)NG * 512;
    unsigned short* XB   = (unsigned short*)(ws + off); off += (size_t)NN * 128 / 2; // bf16 x
    unsigned short* B0   = (unsigned short*)(ws + off); off += (size_t)NN * 512 / 2; // bf16 ping
    unsigned short* B1   = (unsigned short*)(ws + off); off += (size_t)NN * 256 / 2; // bf16 pong
    unsigned short* GINB = (unsigned short*)(ws + off); off += (size_t)NN * 128 / 2; // bf16 gin input
    unsigned short* wbHeap = (unsigned short*)(ws + off);

    // ---- convert weights + x to bf16 ----
    unsigned short* wbPtr = wbHeap;
    auto cvt = [&](const float* src, unsigned short* dst, int n) {
        k_cvt_bf16<<<(n + 255) / 256, 256, 0, stream>>>(src, dst, n);
    };
    auto cvtw = [&](const float* src, int n) {
        unsigned short* dst = wbPtr; wbPtr += n;
        cvt(src, dst, n);
        return (const unsigned short*)dst;
    };
    const unsigned short* wb_lin1 = cvtw(F(IN_LIN1_W), 128 * 512);
    const unsigned short* wb_lin2 = cvtw(F(IN_LIN2_W), 512 * 128);
    const unsigned short* wb_mlp[3][5];
    const float*          bb_mlp[3][5];
    const int mlpBase[3] = { IN_MLP1, IN_MLP2, IN_MLP3 };
    const int din0[3]    = { 128, 32, 32 };
    for (int m = 0; m < 3; ++m) {
        const int dims[6] = { din0[m], 512, 256, 64, 64, 32 };
        for (int l = 0; l < 5; ++l) {
            wb_mlp[m][l] = cvtw(F(mlpBase[m] + 2 * l), dims[l] * dims[l + 1]);
            bb_mlp[m][l] = F(mlpBase[m] + 2 * l + 1);
        }
    }
    cvt(x, XB, NN * 128);

    auto gemm = [&](const unsigned short* A, const unsigned short* Wb, const float* bias,
                    float* Cf, unsigned short* Cb, int N, int K, int M, int act, int outBf) {
        dim3 grid((N + GT_R - 1) / GT_R, (M + GT_C - 1) / GT_C);
        k_gemm_bf16_wmma<<<grid, 256, 0, stream>>>(A, Wb, bias, Cf, Cb, N, K, M, act, outBf);
    };
    auto zero = [&](float* p, size_t n) {
        k_zero_f32<<<(unsigned)((n + 255) / 256), 256, 0, stream>>>(p, n);
    };

    // ---- pre-layers:  h = relu(lin2(sigmoid(lin1(x)))) ----
    gemm(XB, wb_lin1, F(IN_LIN1_B), nullptr, B0, NN, 128, 512, ACT_SIGMOID, 1);
    gemm(B0, wb_lin2, F(IN_LIN2_B), Hf, nullptr, NN, 512, 128, ACT_RELU, 0);   // h in Hf (f32)

    // ---- three GIN layers ----
    const float* layerIn[3] = { Hf, X1, X2 };
    const int    layerD[3]  = { 128, 32, 32 };
    const int    epsIdx[3]  = { IN_EPS1, IN_EPS2, IN_EPS3 };
    const int    bnG[3]     = { IN_BN1_G, IN_BN2_G, IN_BN3_G };
    const int    bnB[3]     = { IN_BN1_B, IN_BN2_B, IN_BN3_B };
    const int    bnAct[3]   = { ACT_SIGMOID, ACT_RELU, ACT_RELU };
    float*       layerOut[3] = { X1, X2, X3 };

    for (int L = 0; L < 3; ++L) {
        const int D = layerD[L];
        const size_t nd = (size_t)NN * D;
        // agg = segment_sum(x[src], dst)  (f32 atomics)
        zero(AG, nd);
        {
            long long t = (long long)EE * (D >> 2);
            k_edge_scatter<<<(unsigned)((t + 255) / 256), 256, 0, stream>>>(layerIn[L], ei, AG, EE, D);
        }
        // gin input = bf16((1+eps)*x + agg) -> GINB
        k_gin_combine_bf<<<(unsigned)((nd + 255) / 256), 256, 0, stream>>>(
            layerIn[L], AG, F(epsIdx[L]), GINB, nd);
        // 5-layer MLP: sigmoid, relu, elu, selu, none; intermediates bf16, final f32
        gemm(GINB, wb_mlp[L][0], bb_mlp[L][0], nullptr, B0, NN, D,   512, ACT_SIGMOID, 1);
        gemm(B0,   wb_mlp[L][1], bb_mlp[L][1], nullptr, B1, NN, 512, 256, ACT_RELU,    1);
        gemm(B1,   wb_mlp[L][2], bb_mlp[L][2], nullptr, B0, NN, 256, 64,  ACT_ELU,     1);
        gemm(B0,   wb_mlp[L][3], bb_mlp[L][3], nullptr, B1, NN, 64,  64,  ACT_SELU,    1);
        gemm(B1,   wb_mlp[L][4], bb_mlp[L][4], Of, nullptr, NN, 64,  32,  ACT_NONE,    0);
        // BatchNorm (training stats, biased var) + activation -> X{L+1} (f32)
        k_bn_stats<<<32, 256, 0, stream>>>(Of, MV, NN, 32);
        k_bn_apply_act<<<(unsigned)(((size_t)NN * 32 + 255) / 256), 256, 0, stream>>>(
            Of, MV, F(bnG[L]), F(bnB[L]), layerOut[L], (size_t)NN * 32, 32, bnAct[L]);
    }

    // ---- global_add_pool of concat(x1,x2,x3) -> G[64,96] ----
    zero(G, (size_t)NG * 96);
    {
        long long t = (long long)NN * 32;
        unsigned nb = (unsigned)((t + 255) / 256);
        k_pool<<<nb, 256, 0, stream>>>(X1, batch, G, NN, 32,  0, 96);
        k_pool<<<nb, 256, 0, stream>>>(X2, batch, G, NN, 32, 32, 96);
        k_pool<<<nb, 256, 0, stream>>>(X3, batch, G, NN, 32, 64, 96);
    }

    // ---- head MLP on [64, *] (tiny; plain kernel) ----
    auto sg = [&](const float* A, int wi, float* C, int N, int K, int M, int act) {
        int n = N * M;
        k_small_gemm<<<(n + 255) / 256, 256, 0, stream>>>(A, F(wi), F(wi + 1), C, N, K, M, act);
    };
    sg(G,  IN_HEAD + 0,  H0, NG,  96, 512, ACT_SIGMOID);
    sg(H0, IN_HEAD + 2,  H1, NG, 512, 256, ACT_RELU);
    sg(H1, IN_HEAD + 4,  H0, NG, 256,  64, ACT_ELU);
    sg(H0, IN_HEAD + 6,  H1, NG,  64,  16, ACT_SELU);
    sg(H1, IN_HEAD + 8,  H0, NG,  16,  16, ACT_GELU);
    sg(H0, IN_HEAD + 10, (float*)d_out, NG, 16, 1, ACT_NONE);
}
